// NodeModel_14628658610613
// MI455X (gfx1250) — matrix operand order repaired and checked by reading, not compiled, not run on hardware
//
#include <hip/hip_runtime.h>
#include <hip/hip_bf16.h>
#include <stdint.h>

// ---------------- problem constants (from reference) ----------------
#define N_NODES   50000
#define N_EDGES   1600000
#define IN_NODE   128
#define EDGE_FEAT 64
#define HIDDEN    128
#define OUT_F     128
#define K1 (IN_NODE + EDGE_FEAT)   // 192
#define K2 HIDDEN                  // 128
#define K3 (IN_NODE + HIDDEN)      // 256
#define K4 HIDDEN                  // 128

// ---------------- WMMA vector types ---------------------------------
typedef __attribute__((ext_vector_type(16))) __bf16         v16bf;
typedef __attribute__((ext_vector_type(8)))  float          v8f;
typedef __attribute__((ext_vector_type(8)))  unsigned short v8u16;
typedef __attribute__((ext_vector_type(16))) unsigned short v16u16;

// f32 -> bf16 round-to-nearest-even via bit manipulation
__device__ __forceinline__ unsigned short f32_to_bf16(float f) {
  unsigned u = __builtin_bit_cast(unsigned, f);
  unsigned r = u + 0x7fffu + ((u >> 16) & 1u);
  return (unsigned short)(r >> 16);
}
__device__ __forceinline__ unsigned pack2bf(float a, float b) {
  return (unsigned)f32_to_bf16(a) | ((unsigned)f32_to_bf16(b) << 16);
}
// Two 16B chunk loads concatenated into one 8-VGPR WMMA fragment.
__device__ __forceinline__ v16bf frag_from(const unsigned short* p0,
                                           const unsigned short* p1) {
  v8u16 lo = *(const v8u16*)p0;
  v8u16 hi = *(const v8u16*)p1;
  v16u16 t = __builtin_shufflevector(lo, hi, 0, 1, 2, 3, 4, 5, 6, 7,
                                     8, 9, 10, 11, 12, 13, 14, 15);
  return __builtin_bit_cast(v16bf, t);
}

// ---------------- helper kernels ------------------------------------
__global__ void zero_ws_kernel(float* p, int n) {
  int i = blockIdx.x * blockDim.x + threadIdx.x;
  if (i < n) p[i] = 0.0f;
}
// dst[n*K + k] = bf16(src[k*N + n])  (weights transposed for B fragments)
__global__ void conv_wT_kernel(const float* __restrict__ src,
                               unsigned short* __restrict__ dst, int K, int N) {
  int i = blockIdx.x * blockDim.x + threadIdx.x;
  if (i < K * N) {
    int k = i / N, n = i % N;
    dst[(size_t)n * K + k] = f32_to_bf16(src[i]);
  }
}
// flat f32 -> bf16 conversion (node features)
__global__ void conv_bf16_kernel(const float* __restrict__ src,
                                 unsigned short* __restrict__ dst, int n) {
  int i = blockIdx.x * blockDim.x + threadIdx.x;
  if (i < n) dst[i] = f32_to_bf16(src[i]);
}

// LDS row strides (elements): padded, rows stay 16B aligned
#define AS1 200   // 192 + 8  (400 B rows)
#define HS1 136   // 128 + 8  (272 B rows)
#define AS2 264   // 256 + 8  (528 B rows)

// ---------------- kernel 1: edge MLP + scatter-add ------------------
// M-blocked: each wave handles 32 edges (two 16-row WMMA subtiles); every B
// fragment feeds two v_wmma issues.  Bias rides in the WMMA C operand via
// tile-invariant cinit vectors (D != C, so no per-tile splat/copies).
__global__ __launch_bounds__(128) void edge_mlp_scatter_kernel(
    const unsigned short* __restrict__ nfbf,   // bf16 node features [N,128]
    const int* __restrict__ edge_index,
    const float* __restrict__ edge_attr,
    const float* __restrict__ b1, const float* __restrict__ b2,
    const unsigned short* __restrict__ w1t, const unsigned short* __restrict__ w2t,
    float* __restrict__ sums, float* __restrict__ counts) {
  __shared__ __align__(16) unsigned short Ast[4][32][AS1];   // 51200 B
  __shared__ __align__(16) unsigned short Hst[4][32][HS1];   // 34816 B
  const int lane = threadIdx.x & 31;
  const int wave = threadIdx.x >> 5;
  const int half = lane >> 4;
  const int lm   = lane & 15;
  const int numTiles = N_EDGES / 32;       // 50000, exact
  const int gw = blockIdx.x * 4 + wave;
  const int nw = gridDim.x * 4;

  // tile-invariant bias-splat C-init vectors (built once, never mutated)
  v8f ci1[8], ci2[8];
#pragma unroll
  for (int nt = 0; nt < 8; ++nt) {
    const float v1 = b1[nt * 16 + lm];
    const float v2 = b2[nt * 16 + lm];
#pragma unroll
    for (int r = 0; r < 8; ++r) { ci1[nt][r] = v1; ci2[nt][r] = v2; }
  }

  for (int tile = gw; tile < numTiles; tile += nw) {
    const int e0 = tile * 32;
    int dstIdx = edge_index[e0 + lane];               // row = destination
    int srcIdx = edge_index[N_EDGES + e0 + lane];     // col = source
    if ((unsigned)dstIdx >= N_NODES) dstIdx = 0;      // defensive clamp
    if ((unsigned)srcIdx >= N_NODES) srcIdx = 0;

    // ---- gather [node_bf16[src] || bf16(edge_attr)] -> 32-row LDS tile ----
#pragma unroll 4
    for (int m = 0; m < 32; ++m) {
      int s = __shfl(srcIdx, m, 32);
      unsigned short* arow = &Ast[wave][m][0];
      *(uint2*)&arow[lane * 4] =
          *(const uint2*)(nfbf + (size_t)s * IN_NODE + lane * 4);
      if (lane < 16) {
        const float4 ea = ((const float4*)(edge_attr + (size_t)(e0 + m) * EDGE_FEAT))[lane];
        *(uint2*)&arow[IN_NODE + lane * 4] =
            make_uint2(pack2bf(ea.x, ea.y), pack2bf(ea.z, ea.w));
      }
    }
    __builtin_prefetch(edge_attr + (size_t)(e0 + 32) * EDGE_FEAT + lane * 16, 0, 1);

    // ---- layer 1: [32x192]x[192x128] (+bias in C), ReLU -> Hst ----
    {
      const unsigned short* ar0 = &Ast[wave][lm][0];
      const unsigned short* ar1 = &Ast[wave][16 + lm][0];
      v16bf a0[K1 / 32], a1[K1 / 32];
#pragma unroll
      for (int kt = 0; kt < K1 / 32; ++kt) {
        a0[kt] = frag_from(ar0 + kt * 32 + 8 * half, ar0 + kt * 32 + 16 + 8 * half);
        a1[kt] = frag_from(ar1 + kt * 32 + 8 * half, ar1 + kt * 32 + 16 + 8 * half);
      }
#pragma unroll
      for (int nt = 0; nt < 8; ++nt) {
        const unsigned short* brow = w1t + (size_t)(nt * 16 + lm) * K1;
        v16bf b0 = frag_from(brow + 8 * half, brow + 16 + 8 * half);
        v8f c0 = __builtin_amdgcn_wmma_f32_16x16x32_bf16(false, a0[0], false, b0, (short)0, ci1[nt], false, false);
        v8f c1 = __builtin_amdgcn_wmma_f32_16x16x32_bf16(false, a1[0], false, b0, (short)0, ci1[nt], false, false);
#pragma unroll
        for (int kt = 1; kt < K1 / 32; ++kt) {
          v16bf b = frag_from(brow + kt * 32 + 8 * half, brow + kt * 32 + 16 + 8 * half);
          c0 = __builtin_amdgcn_wmma_f32_16x16x32_bf16(false, a0[kt], false, b, (short)0, c0, false, false);
          c1 = __builtin_amdgcn_wmma_f32_16x16x32_bf16(false, a1[kt], false, b, (short)0, c1, false, false);
        }
#pragma unroll
        for (int r = 0; r < 8; ++r) {
          Hst[wave][r + 8 * half][nt * 16 + lm]      = f32_to_bf16(fmaxf(c0[r], 0.0f));
          Hst[wave][16 + r + 8 * half][nt * 16 + lm] = f32_to_bf16(fmaxf(c1[r], 0.0f));
        }
      }
    }

    // ---- layer 2: [32x128]x[128x128] (+bias in C), scatter-add ----
    {
      const unsigned short* hr0 = &Hst[wave][lm][0];
      const unsigned short* hr1 = &Hst[wave][16 + lm][0];
      v16bf a0[K2 / 32], a1[K2 / 32];
#pragma unroll
      for (int kt = 0; kt < K2 / 32; ++kt) {
        a0[kt] = frag_from(hr0 + kt * 32 + 8 * half, hr0 + kt * 32 + 16 + 8 * half);
        a1[kt] = frag_from(hr1 + kt * 32 + 8 * half, hr1 + kt * 32 + 16 + 8 * half);
      }
      // hoisted per-row scatter base pointers (nt-invariant)
      float* sp0[8];
      float* sp1[8];
#pragma unroll
      for (int r = 0; r < 8; ++r) {
        int d0 = __shfl(dstIdx, r + 8 * half, 32);
        int d1 = __shfl(dstIdx, 16 + r + 8 * half, 32);
        sp0[r] = sums + (size_t)d0 * HIDDEN + lm;
        sp1[r] = sums + (size_t)d1 * HIDDEN + lm;
      }
#pragma unroll
      for (int nt = 0; nt < 8; ++nt) {
        const unsigned short* brow = w2t + (size_t)(nt * 16 + lm) * K2;
        v16bf b0 = frag_from(brow + 8 * half, brow + 16 + 8 * half);
        v8f c0 = __builtin_amdgcn_wmma_f32_16x16x32_bf16(false, a0[0], false, b0, (short)0, ci2[nt], false, false);
        v8f c1 = __builtin_amdgcn_wmma_f32_16x16x32_bf16(false, a1[0], false, b0, (short)0, ci2[nt], false, false);
#pragma unroll
        for (int kt = 1; kt < K2 / 32; ++kt) {
          v16bf b = frag_from(brow + kt * 32 + 8 * half, brow + kt * 32 + 16 + 8 * half);
          c0 = __builtin_amdgcn_wmma_f32_16x16x32_bf16(false, a0[kt], false, b, (short)0, c0, false, false);
          c1 = __builtin_amdgcn_wmma_f32_16x16x32_bf16(false, a1[kt], false, b, (short)0, c1, false, false);
        }
#pragma unroll
        for (int r = 0; r < 8; ++r) {
          atomicAdd(sp0[r] + nt * 16, c0[r]);
          atomicAdd(sp1[r] + nt * 16, c1[r]);
        }
      }
    }
    atomicAdd(&counts[dstIdx], 1.0f);   // one edge per lane
  }
}

// ---------------- kernel 2: node MLP --------------------------------
__global__ __launch_bounds__(128) void node_mlp_kernel(
    const unsigned short* __restrict__ nfbf,
    const float* __restrict__ b3, const float* __restrict__ b4,
    const unsigned short* __restrict__ w3t, const unsigned short* __restrict__ w4t,
    const float* __restrict__ sums, const float* __restrict__ counts,
    float* __restrict__ out) {
  __shared__ __align__(16) unsigned short Ast[4][16][AS2];
  __shared__ __align__(16) unsigned short Hst[4][16][HS1];
  const int lane = threadIdx.x & 31;
  const int wave = threadIdx.x >> 5;
  const int half = lane >> 4;
  const int lm   = lane & 15;
  const int numTiles = N_NODES / 16;
  const int gw = blockIdx.x * 4 + wave;
  const int nw = gridDim.x * 4;

  v8f ci3[8], ci4[8];
#pragma unroll
  for (int nt = 0; nt < 8; ++nt) {
    const float v3 = b3[nt * 16 + lm];
    const float v4 = b4[nt * 16 + lm];
#pragma unroll
    for (int r = 0; r < 8; ++r) { ci3[nt][r] = v3; ci4[nt][r] = v4; }
  }

  for (int tile = gw; tile < numTiles; tile += nw) {
    const int n0 = tile * 16;
#pragma unroll 4
    for (int m = 0; m < 16; ++m) {
      const int n = n0 + m;
      unsigned short* arow = &Ast[wave][m][0];
      *(uint2*)&arow[lane * 4] =
          *(const uint2*)(nfbf + (size_t)n * IN_NODE + lane * 4);
      const float inv = 1.0f / fmaxf(counts[n], 1.0f);
      float4 sm = ((const float4*)(sums + (size_t)n * HIDDEN))[lane];
      *(uint2*)&arow[IN_NODE + lane * 4] =
          make_uint2(pack2bf(sm.x * inv, sm.y * inv), pack2bf(sm.z * inv, sm.w * inv));
    }

    // ---- layer 3: [16x256]x[256x128] (+bias in C), ReLU -> Hst ----
    const unsigned short* arow = &Ast[wave][lm][0];
    v16bf a3[K3 / 32];
#pragma unroll
    for (int kt = 0; kt < K3 / 32; ++kt)
      a3[kt] = frag_from(arow + kt * 32 + 8 * half, arow + kt * 32 + 16 + 8 * half);
#pragma unroll
    for (int nt = 0; nt < 8; ++nt) {
      const unsigned short* brow = w3t + (size_t)(nt * 16 + lm) * K3;
      v16bf b0 = frag_from(brow + 8 * half, brow + 16 + 8 * half);
      v8f c = __builtin_amdgcn_wmma_f32_16x16x32_bf16(false, a3[0], false, b0, (short)0, ci3[nt], false, false);
#pragma unroll
      for (int kt = 1; kt < K3 / 32; ++kt) {
        v16bf b = frag_from(brow + kt * 32 + 8 * half, brow + kt * 32 + 16 + 8 * half);
        c = __builtin_amdgcn_wmma_f32_16x16x32_bf16(false, a3[kt], false, b, (short)0, c, false, false);
      }
#pragma unroll
      for (int r = 0; r < 8; ++r)
        Hst[wave][r + 8 * half][nt * 16 + lm] = f32_to_bf16(fmaxf(c[r], 0.0f));
    }

    // ---- layer 4: [16x128]x[128x128] (+bias in C) -> out ----
    const unsigned short* hrow = &Hst[wave][lm][0];
    v16bf a4[K4 / 32];
#pragma unroll
    for (int kt = 0; kt < K4 / 32; ++kt)
      a4[kt] = frag_from(hrow + kt * 32 + 8 * half, hrow + kt * 32 + 16 + 8 * half);
#pragma unroll
    for (int nt = 0; nt < 8; ++nt) {
      const unsigned short* brow = w4t + (size_t)(nt * 16 + lm) * K4;
      v16bf b0 = frag_from(brow + 8 * half, brow + 16 + 8 * half);
      v8f c = __builtin_amdgcn_wmma_f32_16x16x32_bf16(false, a4[0], false, b0, (short)0, ci4[nt], false, false);
#pragma unroll
      for (int kt = 1; kt < K4 / 32; ++kt) {
        v16bf b = frag_from(brow + kt * 32 + 8 * half, brow + kt * 32 + 16 + 8 * half);
        c = __builtin_amdgcn_wmma_f32_16x16x32_bf16(false, a4[kt], false, b, (short)0, c, false, false);
      }
#pragma unroll
      for (int r = 0; r < 8; ++r)
        out[(size_t)(n0 + r + 8 * half) * OUT_F + nt * 16 + lm] = c[r];
    }
  }
}

// ---------------- launch --------------------------------------------
extern "C" void kernel_launch(void* const* d_in, const int* in_sizes, int n_in,
                              void* d_out, int out_size, void* d_ws, size_t ws_size,
                              hipStream_t stream) {
  const float* node_feat  = (const float*)d_in[0];
  const int*   edge_index = (const int*)d_in[1];
  const float* edge_attr  = (const float*)d_in[2];
  const float* W1 = (const float*)d_in[3];  const float* b1 = (const float*)d_in[4];
  const float* W2 = (const float*)d_in[5];  const float* b2 = (const float*)d_in[6];
  const float* W3 = (const float*)d_in[7];  const float* b3 = (const float*)d_in[8];
  const float* W4 = (const float*)d_in[9];  const float* b4 = (const float*)d_in[10];
  float* out = (float*)d_out;

  // ws: sums[N,128] f32 | counts[N] f32 | bf16 W^T x4 | bf16 node_feat
  float* sums   = (float*)d_ws;
  float* counts = sums + (size_t)N_NODES * HIDDEN;
  unsigned short* w1t  = (unsigned short*)(counts + N_NODES);
  unsigned short* w2t  = w1t + (size_t)K1 * HIDDEN;
  unsigned short* w3t  = w2t + (size_t)K2 * HIDDEN;
  unsigned short* w4t  = w3t + (size_t)K3 * HIDDEN;
  unsigned short* nfbf = w4t + (size_t)K4 * HIDDEN;   // ~39 MB total ws use

  const int nzero = N_NODES * (HIDDEN + 1);
  zero_ws_kernel<<<(nzero + 255) / 256, 256, 0, stream>>>(sums, nzero);
  conv_wT_kernel<<<(K1 * HIDDEN + 255) / 256, 256, 0, stream>>>(W1, w1t, K1, HIDDEN);
  conv_wT_kernel<<<(K2 * HIDDEN + 255) / 256, 256, 0, stream>>>(W2, w2t, K2, HIDDEN);
  conv_wT_kernel<<<(K3 * HIDDEN + 255) / 256, 256, 0, stream>>>(W3, w3t, K3, HIDDEN);
  conv_wT_kernel<<<(K4 * HIDDEN + 255) / 256, 256, 0, stream>>>(W4, w4t, K4, HIDDEN);
  const int nnf = N_NODES * IN_NODE;
  conv_bf16_kernel<<<(nnf + 255) / 256, 256, 0, stream>>>(node_feat, nfbf, nnf);

  edge_mlp_scatter_kernel<<<2048, 128, 0, stream>>>(nfbf, edge_index, edge_attr,
                                                    b1, b2, w1t, w2t, sums, counts);
  node_mlp_kernel<<<800, 128, 0, stream>>>(nfbf, b3, b4, w3t, w4t, sums, counts, out);
}
